// FourierS2S_9577777070474
// MI455X (gfx1250) — compile-verified, matrix operand order checked by reading, not run on hardware
//
#include <hip/hip_runtime.h>
#include <math.h>

// Problem constants (from reference)
#define S_LEN 128
#define T_LEN 32768
#define KSZ   1024
#define STRD  512
#define NWIN  63
#define HDIM  1024      // HE == HD == C == 1024
#define G4    4096      // 4*H
#define ROWS  (S_LEN * NWIN)   // 8064 = 63 * 128

typedef __attribute__((ext_vector_type(16))) __bf16 v16bf;
typedef __attribute__((ext_vector_type(8)))  float  v8f;
typedef int v4i __attribute__((vector_size(4 * sizeof(int))));

// ---- CDNA5 async global->LDS path (guarded; falls back to plain b128 copies)
#if defined(__HIP_DEVICE_COMPILE__) && defined(__gfx1250__)
#if __has_builtin(__builtin_amdgcn_global_load_async_to_lds_b128)
#define HAS_ASYNC_LDS 1
#endif
#endif

#ifdef HAS_ASYNC_LDS
// builtin signature (from hipcc diagnostic): param0 = v4i __device__* (AS1)
#define AS1P128(p) ((__attribute__((address_space(1))) v4i*)((v4i*)(p)))
#define AS3P128(p) ((__attribute__((address_space(3))) v4i*)((v4i*)(p)))
__device__ __forceinline__ void wait_async0() {
#if __has_builtin(__builtin_amdgcn_s_wait_asynccnt)
  __builtin_amdgcn_s_wait_asynccnt(0);
#else
  asm volatile("s_wait_asynccnt 0" ::: "memory");
#endif
}
#endif

// ---- bf16 helpers (raw ushort storage; round-to-nearest-even convert)
__device__ __forceinline__ float bf2f(unsigned short v) {
  union { unsigned u; float f; } x; x.u = ((unsigned)v) << 16; return x.f;
}
__device__ __forceinline__ unsigned short f2bf(float f) {
  union { float f; unsigned u; } x; x.f = f;
  unsigned r = x.u + 0x7fffu + ((x.u >> 16) & 1u);
  return (unsigned short)(r >> 16);
}
__device__ __forceinline__ float sigf(float x) { return 1.f / (1.f + expf(-x)); }

// =====================================================================
// Generic bf16 WMMA GEMM: C[M,N] = A[M,K] * B[K,N]; fp32 accumulate.
// Ping-pong LDS tiles: async-stage tile k+1 while WMMAs consume tile k.
// Requires: M%128==0, N%64==0, K%32==0, 16B-aligned A/B rows.
// Output: f32 (Cf) and/or bf16 (Cb), same ldc.
// =====================================================================
#define TM 128
#define TN 64
#define TK 32

__global__ __launch_bounds__(256) void k_gemm_bf16(
    const unsigned short* __restrict__ A, int lda,
    const unsigned short* __restrict__ B, int ldb,
    float* __restrict__ Cf, unsigned short* __restrict__ Cb, int ldc,
    int M, int N, int K)
{
  __shared__ unsigned short At[2][TM][TK];       // 2 x 8 KB
  __shared__ unsigned short Bt[2][TN][TK + 2];   // 2 x 4.25 KB, transposed+padded

  const int tid  = threadIdx.x;
  const int wave = tid >> 5;
  const int lane = tid & 31;
  const int m0 = blockIdx.y * TM;
  const int n0 = blockIdx.x * TN;
  const int mlo = lane & 15;
  const int hi  = lane >> 4;

  v8f acc[4];
  for (int i = 0; i < 4; ++i)
    for (int j = 0; j < 8; ++j) acc[i][j] = 0.f;

  // ---- stage A tile 128x32 into buffer b (2 x b128 per thread, async path)
  auto stageA = [&](int k0, int b) {
    #pragma unroll
    for (int s2 = 0; s2 < 2; ++s2) {
      int seg = tid + s2 * 256;            // 512 segments of 8 bf16
      int r   = seg >> 2;
      int sc  = (seg & 3) * 8;
      const unsigned short* g = A + (size_t)(m0 + r) * lda + k0 + sc;
#ifdef HAS_ASYNC_LDS
      __builtin_amdgcn_global_load_async_to_lds_b128(AS1P128((void*)g), AS3P128(&At[b][r][sc]), 0, 0);
#else
      *(uint4*)&At[b][r][sc] = *(const uint4*)g;
#endif
    }
  };
  // ---- stage B tile 32x64, transposed into buffer b (8 elements per thread)
  auto stageB = [&](int k0, int b) {
    int e0 = tid * 8;
    int kk = e0 >> 6;           // 0..31
    int nn = e0 & 63;           // multiple of 8
    const unsigned short* g = B + (size_t)(k0 + kk) * ldb + n0 + nn;
    uint4 d = *(const uint4*)g;
    const unsigned short* ds = (const unsigned short*)&d;
    #pragma unroll
    for (int i = 0; i < 8; ++i) Bt[b][nn + i][kk] = ds[i];
  };

  const int nsteps = K / TK;
  stageA(0, 0);
  stageB(0, 0);
#ifdef HAS_ASYNC_LDS
  wait_async0();
#endif
  __syncthreads();

  for (int s = 0; s < nsteps; ++s) {
    const int cur = s & 1;
    // overlap: stage next tile while computing current one
    if (s + 1 < nsteps) {
      stageA((s + 1) * TK, cur ^ 1);
      stageB((s + 1) * TK, cur ^ 1);
      if (s + 2 < nsteps)   // L2 prefetch hint two tiles ahead
        __builtin_prefetch(A + (size_t)(m0 + (tid >> 1)) * lda + (s + 2) * TK, 0, 1);
    }

    // ---- fragments per ISA 7.12.2 (bf16 16x32 A / 32x16 B layouts)
    union { unsigned u[8]; v16bf v; } fa;
    {
      const unsigned short* ar = &At[cur][wave * 16 + mlo][0];
      #pragma unroll
      for (int p = 0; p < 8; ++p) {
        int kp = ((p < 4) ? (2 * p) : (16 + 2 * (p - 4))) + 8 * hi;
        fa.u[p] = *(const unsigned*)(ar + kp);
      }
    }
    // preload all 4 B fragments into private regs so LDS waits batch up
    unsigned fbu[4][8];
    #pragma unroll
    for (int nt = 0; nt < 4; ++nt) {
      const unsigned short* br = &Bt[cur][nt * 16 + mlo][0];
      #pragma unroll
      for (int p = 0; p < 8; ++p) {
        int kp = ((p < 4) ? (2 * p) : (16 + 2 * (p - 4))) + 8 * hi;
        fbu[nt][p] = *(const unsigned*)(br + kp);
      }
    }
    #pragma unroll
    for (int nt = 0; nt < 4; ++nt) {
      union { unsigned u[8]; v16bf v; } fb;
      #pragma unroll
      for (int p = 0; p < 8; ++p) fb.u[p] = fbu[nt][p];
      acc[nt] = __builtin_amdgcn_wmma_f32_16x16x32_bf16(
          false, fa.v, false, fb.v, (short)0, acc[nt], false, false);
    }
#ifdef HAS_ASYNC_LDS
    wait_async0();
#endif
    __syncthreads();
  }

  // ---- store (C/D layout: VGPR v, lane l -> row v+8*hi, col l&15)
  #pragma unroll
  for (int nt = 0; nt < 4; ++nt) {
    int col = n0 + nt * 16 + mlo;
    #pragma unroll
    for (int v = 0; v < 8; ++v) {
      int row = m0 + wave * 16 + v + 8 * hi;
      float val = acc[nt][v];
      if (Cf) Cf[(size_t)row * ldc + col] = val;
      if (Cb) Cb[(size_t)row * ldc + col] = f2bf(val);
    }
  }
}

// =====================================================================
// Small utility kernels
// =====================================================================
__global__ void k_init(float* h, float* c, unsigned short* hb, unsigned* bar, int n) {
  int i = blockIdx.x * blockDim.x + threadIdx.x;
  if (i < n) { h[i] = 0.f; c[i] = 0.f; hb[i] = 0; }
  if (i == 0) *bar = 0u;
}

// DFT-real matrix: Cm[n*1024+k] = cos(2*pi*n*k/1024) in bf16 (exact range reduction)
__global__ void k_dft(unsigned short* Cm) {
  int i = blockIdx.x * blockDim.x + threadIdx.x;
  int n = i >> 10, k = i & 1023;
  int ph = (n * k) & 1023;
  float ang = 6.283185307179586f * (float)ph * (1.0f / 1024.0f);
  Cm[i] = f2bf(cosf(ang));
}

// Gather strided windows: Aw[(s*63+j)*1024+n] = bf16(x[s*T + j*512 + n])
__global__ void k_gather(const float* __restrict__ x, unsigned short* __restrict__ Aw) {
  int i = blockIdx.x * blockDim.x + threadIdx.x;   // < 8064*1024
  int row = i >> 10, n = i & 1023;
  int s = row / NWIN, j = row % NWIN;
  Aw[i] = f2bf(x[(size_t)s * T_LEN + j * STRD + n]);
}

// Pack/transpose weight: W[R][C] f32 -> Wt[C][R] bf16  (B layout [K][N], ldb=R)
__global__ void k_pack(const float* __restrict__ W, unsigned short* __restrict__ Wt,
                       int R, int C) {
  int i = blockIdx.x * blockDim.x + threadIdx.x;
  if (i >= R * C) return;
  int r = i / C, c = i % C;
  Wt[(size_t)c * R + r] = f2bf(W[i]);
}

// Fused LSTM cell (encoder): gates = gates_hh + Xg[row(s,t)] + b
__global__ void k_lstm_cell(const float* __restrict__ ghh,           // [S][4H]
                            const unsigned short* __restrict__ Xg,   // [S*NW][4H]
                            const float* __restrict__ b,             // [4H]
                            float* __restrict__ c, float* __restrict__ h,
                            unsigned short* __restrict__ hb, int t) {
  int i = blockIdx.x * blockDim.x + threadIdx.x;   // < S*H
  int s = i >> 10, u = i & 1023;
  const float* gr = ghh + (size_t)s * G4;
  const unsigned short* xr = Xg + ((size_t)s * NWIN + t) * G4;
  float gi = gr[u]            + bf2f(xr[u])            + b[u];
  float gf = gr[HDIM + u]     + bf2f(xr[HDIM + u])     + b[HDIM + u];
  float gg = gr[2 * HDIM + u] + bf2f(xr[2 * HDIM + u]) + b[2 * HDIM + u];
  float go = gr[3 * HDIM + u] + bf2f(xr[3 * HDIM + u]) + b[3 * HDIM + u];
  float ii = sigf(gi), ff = sigf(gf), g = tanhf(gg), oo = sigf(go);
  float cn = ff * c[i] + ii * g;
  c[i] = cn;
  float hn = oo * tanhf(cn);
  h[i] = hn;
  hb[i] = f2bf(hn);
}

// Elementwise bias + activation; mode: 0 none, 1 relu, 2 tanh
__global__ void k_bias_act(const float* __restrict__ x, const float* __restrict__ b,
                           float* __restrict__ yf, unsigned short* __restrict__ yb,
                           int N, int total, int mode) {
  int i = blockIdx.x * blockDim.x + threadIdx.x;
  if (i >= total) return;
  float v = x[i] + b[i % N];
  if (mode == 1) v = fmaxf(v, 0.f);
  else if (mode == 2) v = tanhf(v);
  if (yf) yf[i] = v;
  if (yb) yb[i] = f2bf(v);
}

__global__ void k_dec_init(const float* __restrict__ hmid, unsigned short* __restrict__ hbuf) {
  int u = blockIdx.x * blockDim.x + threadIdx.x;   // < 1024
  hbuf[u] = f2bf(hmid[u]);                         // phase 0
}

// =====================================================================
// Persistent decoder: 256 WGs x 256 thr. WG owns 4 hidden units; its 16
// Whh_d columns live in LDS (32KB). One release/acquire global barrier per
// step over a ping-pong h buffer. Cell state lives in registers (tid<4).
// =====================================================================
#define DEC_NWG 256
__global__ __launch_bounds__(256) void k_decoder(
    const unsigned short* __restrict__ Wt,   // WhhD_t [1024][4096]
    const unsigned short* __restrict__ Xg,   // XgD    [8064][4096]
    const float* __restrict__ bd,            // [4096]
    const float* __restrict__ hmid,          // row 0 = h0 (1024 f32)
    unsigned short* __restrict__ hbuf,       // [2][1024] ping-pong
    unsigned short* __restrict__ Hdec,       // [128][1024]
    unsigned* __restrict__ bar)
{
  __shared__ unsigned short wsl[16][1024];   // 32 KB weight slice
  __shared__ unsigned short hs[1024];
  __shared__ float gl[16];

  const int tid = threadIdx.x;
  const int wg  = blockIdx.x;
  const int ubase = wg * 4;

  // Stage weight slice: output o (= gate*4 + unit) -> column n of Wt
  for (int e = tid; e < 16 * 1024; e += 256) {
    int o = e >> 10, k = e & 1023;
    int n = ((o >> 2) << 10) + ubase + (o & 3);
    wsl[o][k] = Wt[(size_t)k * G4 + n];
  }
  const int o   = tid >> 4;
  const int sub = tid & 15;
  const int ncol = ((o >> 2) << 10) + ubase + (o & 3);
  const float bn = (sub == 0) ? bd[ncol] : 0.f;
  float cc = (tid < 4) ? hmid[ubase + tid] : 0.f;   // c0 = h0 (ref uses h for both)
  __syncthreads();

  for (int t = 0; t < ROWS; ++t) {
    const unsigned short* hin = hbuf + (t & 1) * HDIM;
    for (int e = tid; e < HDIM; e += 256) hs[e] = hin[e];
    __syncthreads();

    // 16 lanes per gate output, 64 MACs each
    float p = 0.f;
    const unsigned short* wr = &wsl[o][sub * 64];
    const unsigned short* hr = &hs[sub * 64];
    #pragma unroll 8
    for (int i2 = 0; i2 < 64; ++i2) p = fmaf(bf2f(wr[i2]), bf2f(hr[i2]), p);
    for (int off = 8; off > 0; off >>= 1) p += __shfl_xor(p, off, 16);
    if (sub == 0) gl[o] = p + bf2f(Xg[(size_t)t * G4 + ncol]) + bn;
    __syncthreads();

    if (tid < 4) {   // unit tid: gates at gl[gate*4 + tid]
      float ii = sigf(gl[tid]);
      float ff = sigf(gl[4 + tid]);
      float g  = tanhf(gl[8 + tid]);
      float oo = sigf(gl[12 + tid]);
      cc = ff * cc + ii * g;
      float hv = oo * tanhf(cc);
      unsigned short hb = f2bf(hv);
      hbuf[((t + 1) & 1) * HDIM + ubase + tid] = hb;
      if (t % NWIN == NWIN - 1)
        Hdec[(size_t)(t / NWIN) * HDIM + ubase + tid] = hb;
    }
    __threadfence();
    __syncthreads();
    if (tid == 0) {
      __hip_atomic_fetch_add(bar, 1u, __ATOMIC_RELEASE, __HIP_MEMORY_SCOPE_AGENT);
      unsigned target = (unsigned)(t + 1) * DEC_NWG;
      while (__hip_atomic_load(bar, __ATOMIC_ACQUIRE, __HIP_MEMORY_SCOPE_AGENT) < target)
        __builtin_amdgcn_s_sleep(2);
    }
    __syncthreads();
  }
}

// log_softmax over each row of logits+bo -> out (f32)
__global__ __launch_bounds__(256) void k_logsoftmax(
    const float* __restrict__ z, const float* __restrict__ bo,
    float* __restrict__ out, int C) {
  __shared__ float red[256];
  int r = blockIdx.x, tid = threadIdx.x;
  const float* zr = z + (size_t)r * C;
  float mx = -3.4e38f;
  for (int c = tid; c < C; c += 256) mx = fmaxf(mx, zr[c] + bo[c]);
  red[tid] = mx; __syncthreads();
  for (int s = 128; s > 0; s >>= 1) { if (tid < s) red[tid] = fmaxf(red[tid], red[tid + s]); __syncthreads(); }
  mx = red[0]; __syncthreads();
  float sum = 0.f;
  for (int c = tid; c < C; c += 256) sum += expf(zr[c] + bo[c] - mx);
  red[tid] = sum; __syncthreads();
  for (int s = 128; s > 0; s >>= 1) { if (tid < s) red[tid] += red[tid + s]; __syncthreads(); }
  float lse = mx + logf(red[0]);
  for (int c = tid; c < C; c += 256) out[(size_t)r * C + c] = zr[c] + bo[c] - lse;
}

// =====================================================================
extern "C" void kernel_launch(void* const* d_in, const int* in_sizes, int n_in,
                              void* d_out, int out_size, void* d_ws, size_t ws_size,
                              hipStream_t stream) {
  (void)in_sizes; (void)n_in; (void)out_size; (void)ws_size;
  const float* x    = (const float*)d_in[0];
  const float* WihE = (const float*)d_in[1];
  const float* WhhE = (const float*)d_in[2];
  const float* bE   = (const float*)d_in[3];
  const float* Wm1  = (const float*)d_in[4];
  const float* bm1  = (const float*)d_in[5];
  const float* Wm2  = (const float*)d_in[6];
  const float* bm2  = (const float*)d_in[7];
  const float* WihD = (const float*)d_in[8];
  const float* WhhD = (const float*)d_in[9];
  const float* bD   = (const float*)d_in[10];
  const float* Wo   = (const float*)d_in[11];
  const float* bo   = (const float*)d_in[12];
  float* out = (float*)d_out;

  char* ws = (char*)d_ws;
  size_t off = 0;
  auto alloc = [&](size_t bytes) -> void* {
    void* p = ws + off; off += (bytes + 255) & ~(size_t)255; return p;
  };
  unsigned short* Cmat  = (unsigned short*)alloc((size_t)KSZ * KSZ * 2);
  unsigned short* Awin  = (unsigned short*)alloc((size_t)ROWS * KSZ * 2);
  unsigned short* Fb    = (unsigned short*)alloc((size_t)ROWS * KSZ * 2);
  unsigned short* WihEt = (unsigned short*)alloc((size_t)KSZ * G4 * 2);
  unsigned short* WhhEt = (unsigned short*)alloc((size_t)HDIM * G4 * 2);
  unsigned short* WihDt = (unsigned short*)alloc((size_t)KSZ * G4 * 2);
  unsigned short* WhhDt = (unsigned short*)alloc((size_t)HDIM * G4 * 2);
  unsigned short* Wm1t  = (unsigned short*)alloc((size_t)HDIM * HDIM * 2);
  unsigned short* Wm2t  = (unsigned short*)alloc((size_t)HDIM * HDIM * 2);
  unsigned short* Wot   = (unsigned short*)alloc((size_t)HDIM * HDIM * 2);
  unsigned short* XgE   = (unsigned short*)alloc((size_t)ROWS * G4 * 2);
  unsigned short* XgD   = (unsigned short*)alloc((size_t)ROWS * G4 * 2);
  float* gHH   = (float*)alloc((size_t)S_LEN * G4 * 4);
  float* hE    = (float*)alloc((size_t)S_LEN * HDIM * 4);
  float* cE    = (float*)alloc((size_t)S_LEN * HDIM * 4);
  unsigned short* hEb = (unsigned short*)alloc((size_t)S_LEN * HDIM * 2);
  float* t0f   = (float*)alloc((size_t)S_LEN * HDIM * 4);
  unsigned short* t0b = (unsigned short*)alloc((size_t)S_LEN * HDIM * 2);
  float* hMid  = (float*)alloc((size_t)S_LEN * HDIM * 4);
  unsigned short* hbuf = (unsigned short*)alloc((size_t)2 * HDIM * 2);
  unsigned short* Hdec = (unsigned short*)alloc((size_t)S_LEN * HDIM * 2);
  float* logits = (float*)alloc((size_t)S_LEN * HDIM * 4);
  unsigned* bar = (unsigned*)alloc(256);

  dim3 blk(256);
  auto gemm = [&](const unsigned short* A, int lda, const unsigned short* B, int ldb,
                  float* Cf, unsigned short* Cb, int ldc, int M, int N, int K) {
    dim3 g(N / TN, M / TM);
    k_gemm_bf16<<<g, blk, 0, stream>>>(A, lda, B, ldb, Cf, Cb, ldc, M, N, K);
  };

  // 0) init encoder state + barrier counter (every call: graph-replay safe)
  k_init<<<dim3((S_LEN * HDIM + 255) / 256), blk, 0, stream>>>(hE, cE, hEb, bar, S_LEN * HDIM);
  // 1) DFT-real matrix + window gather + weight packs
  k_dft<<<dim3((KSZ * KSZ) / 256), blk, 0, stream>>>(Cmat);
  k_gather<<<dim3((ROWS * KSZ) / 256), blk, 0, stream>>>(x, Awin);
  k_pack<<<dim3((G4 * KSZ + 255) / 256), blk, 0, stream>>>(WihE, WihEt, G4, KSZ);
  k_pack<<<dim3((G4 * HDIM + 255) / 256), blk, 0, stream>>>(WhhE, WhhEt, G4, HDIM);
  k_pack<<<dim3((G4 * KSZ + 255) / 256), blk, 0, stream>>>(WihD, WihDt, G4, KSZ);
  k_pack<<<dim3((G4 * HDIM + 255) / 256), blk, 0, stream>>>(WhhD, WhhDt, G4, HDIM);
  k_pack<<<dim3((HDIM * HDIM + 255) / 256), blk, 0, stream>>>(Wm1, Wm1t, HDIM, HDIM);
  k_pack<<<dim3((HDIM * HDIM + 255) / 256), blk, 0, stream>>>(Wm2, Wm2t, HDIM, HDIM);
  k_pack<<<dim3((HDIM * HDIM + 255) / 256), blk, 0, stream>>>(Wo,  Wot,  HDIM, HDIM);

  // 2) Fourier features as GEMM: Fb = Awin @ Cmat   (8064x1024x1024)
  gemm(Awin, KSZ, Cmat, KSZ, nullptr, Fb, KSZ, ROWS, KSZ, KSZ);
  // 3) hoisted input projections (no sequential dependence)
  gemm(Fb, KSZ, WihEt, G4, nullptr, XgE, G4, ROWS, G4, KSZ);
  gemm(Fb, KSZ, WihDt, G4, nullptr, XgD, G4, ROWS, G4, KSZ);

  // 4) encoder scan: 63 x (WMMA GEMM hh-term + fused cell)
  for (int t = 0; t < NWIN; ++t) {
    gemm(hEb, HDIM, WhhEt, G4, gHH, nullptr, G4, S_LEN, G4, HDIM);
    k_lstm_cell<<<dim3((S_LEN * HDIM) / 256), blk, 0, stream>>>(gHH, XgE, bE, cE, hE, hEb, t);
  }

  // 5) mid MLP: relu(h@Wm1.T+bm1) @ Wm2.T + bm2 -> tanh
  gemm(hEb, HDIM, Wm1t, HDIM, t0f, nullptr, HDIM, S_LEN, HDIM, HDIM);
  k_bias_act<<<dim3((S_LEN * HDIM) / 256), blk, 0, stream>>>(t0f, bm1, nullptr, t0b, HDIM, S_LEN * HDIM, 1);
  gemm(t0b, HDIM, Wm2t, HDIM, t0f, nullptr, HDIM, S_LEN, HDIM, HDIM);
  k_bias_act<<<dim3((S_LEN * HDIM) / 256), blk, 0, stream>>>(t0f, bm2, hMid, nullptr, HDIM, S_LEN * HDIM, 2);

  // 6) decoder: persistent cooperative kernel (LDS-resident Whh_d slices)
  k_dec_init<<<dim3(HDIM / 256), blk, 0, stream>>>(hMid, hbuf);
  k_decoder<<<dim3(DEC_NWG), blk, 0, stream>>>(WhhDt, XgD, bD, hMid, hbuf, Hdec, bar);

  // 7) output head: logits = Hdec @ Wo.T ; log_softmax(+bo) -> d_out
  gemm(Hdec, HDIM, Wot, HDIM, logits, nullptr, HDIM, S_LEN, HDIM, HDIM);
  k_logsoftmax<<<dim3(S_LEN), blk, 0, stream>>>(logits, bo, out, HDIM);
}
